// DynamicTemporalAttention_18253611008848
// MI455X (gfx1250) — matrix operand order repaired and checked by reading, not compiled, not run on hardware
//
#include <hip/hip_runtime.h>
#include <math.h>
#include <stdint.h>

#define BD   2
#define SEQ  2048
#define DM   1024
#define NH   16
#define HD   64
#define NTOK (BD*SEQ)   // 4096

typedef __attribute__((ext_vector_type(16))) __bf16 v16bf;
typedef __attribute__((ext_vector_type(8)))  float  v8f;
typedef __attribute__((ext_vector_type(8)))  unsigned short u16x8;
typedef __attribute__((ext_vector_type(4)))  unsigned int u32x4;
typedef __attribute__((ext_vector_type(8)))  int i32x8;
typedef __attribute__((ext_vector_type(4)))  int i32x4;

// ---------- helpers ----------

__device__ __forceinline__ unsigned short f2bf(float f) {
    unsigned u = __builtin_bit_cast(unsigned, f);
    u += 0x7fffu + ((u >> 16) & 1u);          // round-to-nearest-even
    return (unsigned short)(u >> 16);
}

union FragU { v16bf v; u16x8 h[2]; };

// A fragment: 16x32 bf16, source row-major with leading dim `ld` (elements).
__device__ __forceinline__ v16bf load_frag_a(const unsigned short* p, int ld) {
    int lane = threadIdx.x & 31;
    int r = lane & 15, g = lane >> 4;
    FragU u;
    u.h[0] = *(const u16x8*)(p + r * ld + g * 8);
    u.h[1] = *(const u16x8*)(p + r * ld + g * 8 + 16);
    return u.v;
}

// B fragment: 32x16 bf16 (KxN), source laid out [col][k] with leading dim `ld`.
__device__ __forceinline__ v16bf load_frag_b(const unsigned short* p, int ld) {
    int lane = threadIdx.x & 31;
    int c = lane & 15, g = lane >> 4;
    FragU u;
    u.h[0] = *(const u16x8*)(p + c * ld + g * 8);
    u.h[1] = *(const u16x8*)(p + c * ld + g * 8 + 16);
    return u.v;
}

__device__ __forceinline__ v8f wmma_bf16(v16bf a, v16bf b, v8f c) {
    return __builtin_amdgcn_wmma_f32_16x16x32_bf16(false, a, false, b, (short)0, c, false, false);
}

// async global -> LDS (16B per lane), tracked by ASYNCcnt
__device__ __forceinline__ void async_ld_b128(unsigned short* ldsp,
                                              const unsigned short* gp) {
    unsigned loff = (unsigned)(uintptr_t)ldsp;   // low 32 bits of generic = LDS offset
    asm volatile("global_load_async_to_lds_b128 %0, %1, off"
                 :: "v"(loff), "v"(gp) : "memory");
}
__device__ __forceinline__ void wait_async() {
    asm volatile("s_wait_asynccnt 0x0" ::: "memory");
}

// ---------- kernel 0: f32 -> bf16 cast ----------

__global__ void dta_cast_bf16(const float* __restrict__ in,
                              unsigned short* __restrict__ out, int n) {
    int i = blockIdx.x * blockDim.x + threadIdx.x;
    int stride = gridDim.x * blockDim.x;
    for (; i < n; i += stride) out[i] = f2bf(in[i]);
}

// ---------- kernel 1: fused QKV projection ----------
// grid: (NTOK/128, DM/64, 3), block 256 (8 waves). Each wave: 32x32 outputs.
// Double-buffered LDS; A tiles staged with async-to-LDS.

__global__ __launch_bounds__(256) void dta_qkv_gemm(
    const unsigned short* __restrict__ xb,
    const unsigned short* __restrict__ wq, const unsigned short* __restrict__ wk,
    const unsigned short* __restrict__ wv,
    const float* __restrict__ bq, const float* __restrict__ bk,
    const float* __restrict__ bv,
    unsigned short* __restrict__ oq, unsigned short* __restrict__ ok,
    unsigned short* __restrict__ ov)
{
    const unsigned short* w = (blockIdx.z == 0) ? wq : (blockIdx.z == 1) ? wk : wv;
    const float* bias       = (blockIdx.z == 0) ? bq : (blockIdx.z == 1) ? bk : bv;
    unsigned short* out     = (blockIdx.z == 0) ? oq : (blockIdx.z == 1) ? ok : ov;

    __shared__ unsigned short lA[2][128 * 32];   // x tiles, row-major
    __shared__ unsigned short lB[2][64 * 32];    // W tiles, transposed [col][k]

    int t = threadIdx.x, lane = t & 31, wave = t >> 5;
    int wm = wave & 3, wn = wave >> 2;           // 4 x 2 wave grid
    int m0 = blockIdx.x * 128, n0 = blockIdx.y * 64;

    auto stageA = [&](int kk, int buf) {         // 128 rows x 32 k, async
        int row = t >> 1, half = t & 1;
        const unsigned short* src = xb + (size_t)(m0 + row) * DM + kk + half * 16;
        unsigned short* dst = &lA[buf][row * 32 + half * 16];
        async_ld_b128(dst, src);
        async_ld_b128(dst + 8, src + 8);
    };
    auto stageB = [&](int kk, int buf) {         // 32 k x 64 n -> lB[n][k]
        int k = t >> 3, cg = t & 7;
        u16x8 v = *(const u16x8*)(w + (size_t)(kk + k) * DM + n0 + cg * 8);
        #pragma unroll
        for (int i = 0; i < 8; ++i) lB[buf][(cg * 8 + i) * 32 + k] = v[i];
    };

    v8f zero = {0.f,0.f,0.f,0.f,0.f,0.f,0.f,0.f};
    v8f acc[2][2];
    acc[0][0] = zero; acc[0][1] = zero; acc[1][0] = zero; acc[1][1] = zero;

    stageA(0, 0);
    stageB(0, 0);
    wait_async();
    __syncthreads();

    int cur = 0;
    for (int kk = 0; kk < DM; kk += 32) {
        int nxt = cur ^ 1;
        bool more = (kk + 32 < DM);
        if (more) {                      // stage next tile while computing this one
            stageA(kk + 32, nxt);
            stageB(kk + 32, nxt);
            if (kk + 64 < DM)
                __builtin_prefetch(w + (size_t)(kk + 64 + (t >> 3)) * DM + n0, 0, 1);
        }
        v16bf a0 = load_frag_a(&lA[cur][(wm * 32     ) * 32], 32);
        v16bf a1 = load_frag_a(&lA[cur][(wm * 32 + 16) * 32], 32);
        v16bf b0 = load_frag_b(&lB[cur][(wn * 32     ) * 32], 32);
        v16bf b1 = load_frag_b(&lB[cur][(wn * 32 + 16) * 32], 32);
        acc[0][0] = wmma_bf16(a0, b0, acc[0][0]);
        acc[0][1] = wmma_bf16(a0, b1, acc[0][1]);
        acc[1][0] = wmma_bf16(a1, b0, acc[1][0]);
        acc[1][1] = wmma_bf16(a1, b1, acc[1][1]);
        if (more) wait_async();
        __syncthreads();
        cur = nxt;
    }

    int g = lane >> 4, c = lane & 15;
    #pragma unroll
    for (int mi = 0; mi < 2; ++mi)
        #pragma unroll
        for (int ni = 0; ni < 2; ++ni) {
            int coln = n0 + wn * 32 + ni * 16 + c;
            float bvv = bias[coln];
            int hh = coln >> 6, dd = coln & 63;
            #pragma unroll
            for (int j = 0; j < 8; ++j) {
                int tok = m0 + wm * 32 + mi * 16 + g * 8 + j;
                int bb = tok >> 11, ss = tok & (SEQ - 1);
                out[(((size_t)bb * NH + hh) * SEQ + ss) * HD + dd] =
                    f2bf(acc[mi][ni][j] + bvv);
            }
        }
}

// ---------- kernel 2: gated attention (two-pass flash) ----------
// grid: (SEQ/64, B*H), block 128 (4 waves); wave w owns 16 query rows.
// K tiles staged by the Tensor Data Mover (contiguous 4KB row block).

__global__ __launch_bounds__(128) void dta_attn(
    const unsigned short* __restrict__ qb, const unsigned short* __restrict__ kb,
    const unsigned short* __restrict__ vb,
    const float* __restrict__ lthr, const float* __restrict__ gshp,
    unsigned short* __restrict__ ob)
{
    __shared__ unsigned short lK[32 * 64];     // K tile, row-major [key][d]
    __shared__ unsigned short lVt[64 * 32];    // V tile transposed [d][key]
    __shared__ unsigned short lW[4][16 * 32];  // per-wave gated-weight tile

    int t = threadIdx.x, lane = t & 31, wave = t >> 5;
    int bh = blockIdx.y;
    int b = bh >> 4, h = bh & 15;
    const unsigned short* qp = qb + (size_t)bh * SEQ * HD;
    const unsigned short* kp = kb + (size_t)bh * SEQ * HD;
    const unsigned short* vp = vb + (size_t)bh * SEQ * HD;
    int q0 = blockIdx.x * 64 + wave * 16;

    // Q A-fragments (16 x 64 split into two K=32 chunks) kept in registers.
    v16bf aq0 = load_frag_a(qp + (size_t)q0 * HD,      HD);
    v16bf aq1 = load_frag_a(qp + (size_t)q0 * HD + 32, HD);

    float thr = 1.0f / (1.0f + __expf(-lthr[h]));
    float shp = gshp[h];
    const float sc = 0.125f;   // 1/sqrt(64)
    int g = lane >> 4, c = lane & 15;
    v8f zero = {0.f,0.f,0.f,0.f,0.f,0.f,0.f,0.f};

    // TDM: copy 32 keys x 64 d (4KB, contiguous) into lK. Wave 0 issues.
    auto stageK_tdm = [&](int kb0) {
        if (wave == 0) {
            unsigned lds = (unsigned)(uintptr_t)(&lK[0]);
            unsigned long long ga =
                (unsigned long long)(uintptr_t)(kp + (size_t)kb0 * HD);
            // D# group0: count=1 | lds_addr | global_addr(57b) | type=2
            u32x4 g0 = { 1u, lds, (unsigned)ga,
                         ((unsigned)((ga >> 32) & 0x01ffffffull)) | 0x80000000u };
            // D# group1: data_size=2B; tensor_dim0=2048; tensor_dim1=1;
            //            tile_dim0=2048 (1-D copy); tensor_dim0_stride=2048
            i32x8 g1 = { 0x00010000, 0x08000000, 0x00010000, 0x08000000,
                         0, 2048, 0, 0 };
            i32x4 gz4 = { 0, 0, 0, 0 };
            i32x8 gz8 = { 0, 0, 0, 0, 0, 0, 0, 0 };
            __builtin_amdgcn_tensor_load_to_lds(g0, g1, gz4, gz4, gz8, 0);
            __builtin_amdgcn_s_wait_tensorcnt(0);
        }
    };
    auto stageV = [&](int kb0) {               // transpose -> lVt[d][key]
        int key = t & 31, dseg = t >> 5;
        u16x8 v0 = *(const u16x8*)(vp + (size_t)(kb0 + key) * HD + dseg * 16);
        u16x8 v1 = *(const u16x8*)(vp + (size_t)(kb0 + key) * HD + dseg * 16 + 8);
        #pragma unroll
        for (int i = 0; i < 8; ++i) lVt[(dseg * 16 + i) * 32 + key] = v0[i];
        #pragma unroll
        for (int i = 0; i < 8; ++i) lVt[(dseg * 16 + 8 + i) * 32 + key] = v1[i];
    };
    auto scores = [&](int kt) -> v8f {
        v16bf b0 = load_frag_b(lK + (kt * 16) * 64,      64);   // d 0..31
        v16bf b1 = load_frag_b(lK + (kt * 16) * 64 + 32, 64);   // d 32..63
        v8f s = wmma_bf16(aq0, b0, zero);
        s = wmma_bf16(aq1, b1, s);
        v8f r;
        #pragma unroll
        for (int j = 0; j < 8; ++j) r[j] = s[j] * sc;
        return r;
    };

    // ---- pass 1: row max + softmax denominator (online) ----
    float mrun[8], lrun[8];
    #pragma unroll
    for (int j = 0; j < 8; ++j) { mrun[j] = -1e30f; lrun[j] = 0.f; }

    for (int kb0 = 0; kb0 < SEQ; kb0 += 32) {
        __syncthreads();
        stageK_tdm(kb0);
        __syncthreads();
        v8f s0 = scores(0), s1 = scores(1);
        #pragma unroll
        for (int j = 0; j < 8; ++j) {
            float mx = fmaxf(s0[j], s1[j]);
            mx = fmaxf(mx, __shfl_xor(mx, 1));
            mx = fmaxf(mx, __shfl_xor(mx, 2));
            mx = fmaxf(mx, __shfl_xor(mx, 4));
            mx = fmaxf(mx, __shfl_xor(mx, 8));
            float mnew = fmaxf(mrun[j], mx);
            float fac  = __expf(mrun[j] - mnew);
            float ps   = __expf(s0[j] - mnew) + __expf(s1[j] - mnew);
            ps += __shfl_xor(ps, 1);
            ps += __shfl_xor(ps, 2);
            ps += __shfl_xor(ps, 4);
            ps += __shfl_xor(ps, 8);
            lrun[j] = lrun[j] * fac + ps;
            mrun[j] = mnew;
        }
    }
    float invl[8];
    #pragma unroll
    for (int j = 0; j < 8; ++j) invl[j] = 1.0f / lrun[j];

    // ---- pass 2: gated probabilities @ V ----
    v8f o[4];
    #pragma unroll
    for (int i = 0; i < 4; ++i) o[i] = zero;
    unsigned short* lw = lW[wave];

    for (int kb0 = 0; kb0 < SEQ; kb0 += 32) {
        __syncthreads();
        stageK_tdm(kb0);
        stageV(kb0);
        __syncthreads();
        v8f s0 = scores(0), s1 = scores(1);
        #pragma unroll
        for (int j = 0; j < 8; ++j) {
            float p0 = __expf(s0[j] - mrun[j]) * invl[j];
            float p1 = __expf(s1[j] - mrun[j]) * invl[j];
            float w0 = p0 / (1.0f + __expf(-shp * (p0 - thr)));
            float w1 = p1 / (1.0f + __expf(-shp * (p1 - thr)));
            int row = g * 8 + j;
            lw[row * 32 + c]      = f2bf(w0);
            lw[row * 32 + 16 + c] = f2bf(w1);
        }
        __syncthreads();                 // C-layout -> A-layout via LDS
        v16bf aw = load_frag_a(lw, 32);
        #pragma unroll
        for (int ni = 0; ni < 4; ++ni) {
            v16bf bv = load_frag_b(lVt + (ni * 16) * 32, 32);
            o[ni] = wmma_bf16(aw, bv, o[ni]);
        }
    }

    // write [B, S, D] bf16 (col = h*64 + d) for the output projection
    #pragma unroll
    for (int ni = 0; ni < 4; ++ni)
        #pragma unroll
        for (int j = 0; j < 8; ++j) {
            int m = g * 8 + j;
            size_t row = (size_t)b * SEQ + q0 + m;
            ob[row * DM + h * HD + ni * 16 + c] = f2bf(o[ni][j]);
        }
}

// ---------- kernel 3: output projection (f32 result) ----------

__global__ __launch_bounds__(256) void dta_out_gemm(
    const unsigned short* __restrict__ ab, const unsigned short* __restrict__ w,
    const float* __restrict__ bias, float* __restrict__ out)
{
    __shared__ unsigned short lA[2][128 * 32];
    __shared__ unsigned short lB[2][64 * 32];

    int t = threadIdx.x, lane = t & 31, wave = t >> 5;
    int wm = wave & 3, wn = wave >> 2;
    int m0 = blockIdx.x * 128, n0 = blockIdx.y * 64;

    auto stageA = [&](int kk, int buf) {
        int row = t >> 1, half = t & 1;
        const unsigned short* src = ab + (size_t)(m0 + row) * DM + kk + half * 16;
        unsigned short* dst = &lA[buf][row * 32 + half * 16];
        async_ld_b128(dst, src);
        async_ld_b128(dst + 8, src + 8);
    };
    auto stageB = [&](int kk, int buf) {
        int k = t >> 3, cg = t & 7;
        u16x8 v = *(const u16x8*)(w + (size_t)(kk + k) * DM + n0 + cg * 8);
        #pragma unroll
        for (int i = 0; i < 8; ++i) lB[buf][(cg * 8 + i) * 32 + k] = v[i];
    };

    v8f zero = {0.f,0.f,0.f,0.f,0.f,0.f,0.f,0.f};
    v8f acc[2][2];
    acc[0][0] = zero; acc[0][1] = zero; acc[1][0] = zero; acc[1][1] = zero;

    stageA(0, 0);
    stageB(0, 0);
    wait_async();
    __syncthreads();

    int cur = 0;
    for (int kk = 0; kk < DM; kk += 32) {
        int nxt = cur ^ 1;
        bool more = (kk + 32 < DM);
        if (more) {
            stageA(kk + 32, nxt);
            stageB(kk + 32, nxt);
            if (kk + 64 < DM)
                __builtin_prefetch(w + (size_t)(kk + 64 + (t >> 3)) * DM + n0, 0, 1);
        }
        v16bf a0 = load_frag_a(&lA[cur][(wm * 32     ) * 32], 32);
        v16bf a1 = load_frag_a(&lA[cur][(wm * 32 + 16) * 32], 32);
        v16bf b0 = load_frag_b(&lB[cur][(wn * 32     ) * 32], 32);
        v16bf b1 = load_frag_b(&lB[cur][(wn * 32 + 16) * 32], 32);
        acc[0][0] = wmma_bf16(a0, b0, acc[0][0]);
        acc[0][1] = wmma_bf16(a0, b1, acc[0][1]);
        acc[1][0] = wmma_bf16(a1, b0, acc[1][0]);
        acc[1][1] = wmma_bf16(a1, b1, acc[1][1]);
        if (more) wait_async();
        __syncthreads();
        cur = nxt;
    }

    int g = lane >> 4, c = lane & 15;
    #pragma unroll
    for (int mi = 0; mi < 2; ++mi)
        #pragma unroll
        for (int ni = 0; ni < 2; ++ni) {
            int coln = n0 + wn * 32 + ni * 16 + c;
            float bvv = bias[coln];
            #pragma unroll
            for (int j = 0; j < 8; ++j) {
                int tok = m0 + wm * 32 + mi * 16 + g * 8 + j;
                out[(size_t)tok * DM + coln] = acc[mi][ni][j] + bvv;
            }
        }
}

// ---------- launch ----------

extern "C" void kernel_launch(void* const* d_in, const int* in_sizes, int n_in,
                              void* d_out, int out_size, void* d_ws, size_t ws_size,
                              hipStream_t stream) {
    const float* x    = (const float*)d_in[0];
    const float* wq_w = (const float*)d_in[1];
    const float* wq_b = (const float*)d_in[2];
    const float* wk_w = (const float*)d_in[3];
    const float* wk_b = (const float*)d_in[4];
    const float* wv_w = (const float*)d_in[5];
    const float* wv_b = (const float*)d_in[6];
    const float* wo_w = (const float*)d_in[7];
    const float* wo_b = (const float*)d_in[8];
    const float* lth  = (const float*)d_in[9];
    const float* gsh  = (const float*)d_in[10];

    char* ws = (char*)d_ws;
    unsigned short* xb  = (unsigned short*)(ws + 0);          //  8 MB
    unsigned short* wqb = (unsigned short*)(ws + 8388608);    //  2 MB
    unsigned short* wkb = (unsigned short*)(ws + 10485760);
    unsigned short* wvb = (unsigned short*)(ws + 12582912);
    unsigned short* wob = (unsigned short*)(ws + 14680064);
    unsigned short* qbf = (unsigned short*)(ws + 16777216);   //  8 MB, [B,H,S,HD]
    unsigned short* kbf = (unsigned short*)(ws + 25165824);
    unsigned short* vbf = (unsigned short*)(ws + 33554432);
    unsigned short* obf = (unsigned short*)(ws + 41943040);   //  8 MB, [B,S,D]

    dta_cast_bf16<<<4096, 256, 0, stream>>>(x, xb, NTOK * DM);
    dta_cast_bf16<<<1024, 256, 0, stream>>>(wq_w, wqb, DM * DM);
    dta_cast_bf16<<<1024, 256, 0, stream>>>(wk_w, wkb, DM * DM);
    dta_cast_bf16<<<1024, 256, 0, stream>>>(wv_w, wvb, DM * DM);
    dta_cast_bf16<<<1024, 256, 0, stream>>>(wo_w, wob, DM * DM);

    dta_qkv_gemm<<<dim3(NTOK / 128, DM / 64, 3), 256, 0, stream>>>(
        xb, wqb, wkb, wvb, wq_b, wk_b, wv_b, qbf, kbf, vbf);

    dta_attn<<<dim3(SEQ / 64, BD * NH), 128, 0, stream>>>(
        qbf, kbf, vbf, lth, gsh, obf);

    dta_out_gemm<<<dim3(NTOK / 128, DM / 64), 256, 0, stream>>>(
        obf, wob, wo_b, (float*)d_out);
}
